// DLI_loss_3_6614249636353
// MI455X (gfx1250) — matrix operand order repaired and checked by reading, not compiled, not run on hardware
//
#include <hip/hip_runtime.h>
#include <hip/hip_bf16.h>

typedef __bf16 bf16;
typedef __attribute__((ext_vector_type(16))) __bf16 v16bf;
typedef __attribute__((ext_vector_type(8)))  __bf16 bf16x8;
typedef __attribute__((ext_vector_type(4)))  __bf16 bf16x4;
typedef __attribute__((ext_vector_type(8)))  float  v8f;
typedef __attribute__((ext_vector_type(4)))  float  f32x4;

#define BSZ   32
#define SEQ   2048
#define TN    32
#define ENC   512
#define HID   512
#define G4    2048         // 4*HID
#define JW    29           // TN-3
#define NW    (BSZ*JW)     // 928 windows
#define MROWS (BSZ*TN)     // 1024 turn rows

// ---------------- 1. ragged segment-mean pooling -> turn_states ----------------
// grid (BSZ, TN), block 128: each thread owns one float4 column group.
// Encoder is streamed once (134 MB) -> non-temporal loads so it does not evict
// the GEMM working set (weights/XG, ~15 MB) from the 192 MB L2.
__global__ void pool_kernel(const float* __restrict__ enc,
                            const int*   __restrict__ end_ids,
                            float*       __restrict__ ts) {
  int b = blockIdx.x;
  int t = blockIdx.y;
  int start = (t == 0) ? 0 : (end_ids[b*TN + t - 1] + 1);
  int end   = end_ids[b*TN + t];
  int cnt   = end - start + 1;
  float inv = 1.0f / (float)cnt;
  int c4 = threadIdx.x;                         // 128 float4 columns
  const f32x4* base = (const f32x4*)(enc + ((size_t)b*SEQ + start) * ENC);
  f32x4 acc = {0.f, 0.f, 0.f, 0.f};
  for (int k = 0; k < cnt; ++k) {
    f32x4 v = __builtin_nontemporal_load(base + (size_t)k*(ENC/4) + c4);
    acc += v;
  }
  acc *= inv;
  f32x4* out = (f32x4*)(ts + ((size_t)b*TN + t) * ENC);
  out[c4] = acc;
}

// ---------------- 2. f32 -> bf16 conversion (4-wide) -------------------------
__global__ void f2bf_kernel(const float* __restrict__ src, bf16* __restrict__ dst, int n4) {
  int i = blockIdx.x * blockDim.x + threadIdx.x;
  if (i < n4) {
    f32x4 v = *((const f32x4*)src + i);
    bf16x4 o;
    o[0] = (bf16)v[0]; o[1] = (bf16)v[1]; o[2] = (bf16)v[2]; o[3] = (bf16)v[3];
    *((bf16x4*)dst + i) = o;
  }
}

// ---------------- 3. shared input-gate GEMM: XG = TS @ W_ih^T + (b_ih+b_hh) ----
// One wave (32 threads) per 16x16 output tile; K=512 -> 16 WMMAs.
// A (16x32 bf16): row = lane&15, K = (e>>3)*16 + (lane>>4)*8 + (e&7)
// B (32x16 bf16): col(gate) = lane&15, K = (e>>3)*16 + (lane>>4)*8 + (e&7)
__global__ void xgemm_kernel(const bf16* __restrict__ A,     // [1024,512]
                             const bf16* __restrict__ W,     // [2048,512]
                             const float* __restrict__ bih,
                             const float* __restrict__ bhh,
                             float* __restrict__ XG) {       // [1024,2048]
  int n0   = blockIdx.x * 16;
  int m0   = blockIdx.y * 16;
  int lane = threadIdx.x;
  int half = lane >> 4;
  int l15  = lane & 15;
  int col  = n0 + l15;
  float bias = bih[col] + bhh[col];
  const bf16* arow = A + (size_t)(m0 + l15) * HID;
  const bf16* brow = W + (size_t)col * HID;
  v8f acc = {};
  for (int k0 = 0; k0 < HID; k0 += 32) {
    v16bf a, bm;
    bf16x8 alo = *(const bf16x8*)(arow + k0 + half*8);
    bf16x8 ahi = *(const bf16x8*)(arow + k0 + 16 + half*8);
    bf16x8 blo = *(const bf16x8*)(brow + k0 + half*8);
    bf16x8 bhi = *(const bf16x8*)(brow + k0 + 16 + half*8);
#pragma unroll
    for (int i = 0; i < 8; ++i) { a[i]=alo[i]; a[8+i]=ahi[i]; bm[i]=blo[i]; bm[8+i]=bhi[i]; }
    acc = __builtin_amdgcn_wmma_f32_16x16x32_bf16(false, a, false, bm, (short)0, acc, false, false);
  }
#pragma unroll
  for (int r = 0; r < 8; ++r) {
    int m = r + half*8;                        // C/D layout: VGPR r -> M = r (+8 for hi lanes)
    XG[(size_t)(m0 + m)*G4 + col] = acc[r] + bias;
  }
}

// ---------------- 4. fused LSTM recurrence + sscore -------------------------
// One block per 16-window tile (58 blocks). h,c,G live in LDS (192 KB < 320 KB WGP).
// Steps 1,2: G = h @ W_hh^T via WMMA (8 waves x 16 n-tiles); step 0 reads XG directly.
__global__ void lstm_kernel(const float* __restrict__ XG,    // [1024,2048]
                            const bf16*  __restrict__ Whh,   // [2048,512] bf16
                            const float* __restrict__ fcw,   // [1024], src part = [0:512)
                            float*       __restrict__ sscore) { // [928]
  extern __shared__ float smem[];
  float* h  = smem;               // 16*512
  float* c  = h + 16*HID;         // 16*512
  float* Gm = c + 16*HID;         // 16*2048
  int tid  = threadIdx.x;         // 256 = 8 waves
  int wave = tid >> 5;
  int lane = tid & 31;
  int half = lane >> 4;
  int l15  = lane & 15;
  int wb   = blockIdx.x * 16;     // window tile base

  for (int i = tid; i < 16*HID; i += 256) { h[i] = 0.f; c[i] = 0.f; }
  __syncthreads();

  for (int step = 0; step < 3; ++step) {
    if (step > 0) {
      for (int ntl = 0; ntl < 16; ++ntl) {
        int n0  = (wave*16 + ntl) * 16;
        int col = n0 + l15;
        const bf16* brow = Whh + (size_t)col * HID;
        v8f acc = {};
        for (int k0 = 0; k0 < HID; k0 += 32) {
          v16bf a, bm;
#pragma unroll
          for (int e = 0; e < 16; ++e) {
            int kk = ((e >> 3) * 16) + half*8 + (e & 7);
            a[e] = (bf16)h[l15*HID + k0 + kk];
          }
          bf16x8 blo = *(const bf16x8*)(brow + k0 + half*8);
          bf16x8 bhi = *(const bf16x8*)(brow + k0 + 16 + half*8);
#pragma unroll
          for (int i = 0; i < 8; ++i) { bm[i]=blo[i]; bm[8+i]=bhi[i]; }
          acc = __builtin_amdgcn_wmma_f32_16x16x32_bf16(false, a, false, bm, (short)0, acc, false, false);
        }
#pragma unroll
        for (int r = 0; r < 8; ++r) {
          int m = r + half*8;
          int w = wb + m; int b = w / JW; int j = w % JW;
          int xrow = b*TN + j + step;
          Gm[m*G4 + col] = acc[r] + XG[(size_t)xrow*G4 + col];
        }
      }
    }
    __syncthreads();
    // elementwise LSTM cell update (pytorch gate order i,f,g,o)
    for (int i = tid; i < 16*HID; i += 256) {
      int m  = i >> 9;
      int hc = i & 511;
      float gi, gf, gg, go;
      if (step == 0) {
        int w = wb + m; int b = w / JW; int j = w % JW;
        const float* xr = XG + (size_t)(b*TN + j)*G4;
        gi = xr[hc]; gf = xr[HID+hc]; gg = xr[2*HID+hc]; go = xr[3*HID+hc];
      } else {
        const float* gr = Gm + m*G4;
        gi = gr[hc]; gf = gr[HID+hc]; gg = gr[2*HID+hc]; go = gr[3*HID+hc];
      }
      float si = 1.f/(1.f+__expf(-gi));
      float sf = 1.f/(1.f+__expf(-gf));
      float so = 1.f/(1.f+__expf(-go));
      float cn = sf * c[i] + si * tanhf(gg);
      c[i] = cn;
      h[i] = so * tanhf(cn);
    }
    __syncthreads();
  }
  if (tid < 16) {
    float s = 0.f;
    for (int k = 0; k < HID; ++k) s += h[tid*HID + k] * fcw[k];
    sscore[wb + tid] = s;
  }
}

// ---------------- 5. tscore + masked logsumexp loss + mean -------------------
__global__ void final_kernel(const float* __restrict__ ts,     // [1024,512]
                             const float* __restrict__ fcw,    // [1024], tgt part = [512:1024)
                             const float* __restrict__ fcb,    // [1]
                             const float* __restrict__ sscore, // [928]
                             float* __restrict__ out) {
  __shared__ float tsc[MROWS];
  __shared__ float red[1024];
  int tid = threadIdx.x;
  {
    const float* row = ts + (size_t)tid * ENC;
    float s = 0.f;
    for (int k = 0; k < HID; ++k) s += row[k] * fcw[HID + k];
    tsc[tid] = s;
  }
  __syncthreads();
  float loss = 0.f;
  if (tid < NW) {
    int b = tid / JW, j = tid % JW;
    float sc = sscore[tid] + fcb[0];
    int nk = JW - j;                            // valid candidates: j+3+k < 32
    const float* tb = &tsc[b*TN + j + 3];
    float v0 = sc + tb[0];
    float mx = v0;
    for (int k = 1; k < nk; ++k) mx = fmaxf(mx, sc + tb[k]);
    float sum = 0.f;
    for (int k = 0; k < nk; ++k) sum += __expf(sc + tb[k] - mx);
    loss = mx + __logf(sum) - v0;
  }
  red[tid] = loss;
  __syncthreads();
  for (int s = 512; s > 0; s >>= 1) {
    if (tid < s) red[tid] += red[tid + s];
    __syncthreads();
  }
  if (tid == 0) out[0] = red[0] / (float)NW;
}

extern "C" void kernel_launch(void* const* d_in, const int* in_sizes, int n_in,
                              void* d_out, int out_size, void* d_ws, size_t ws_size,
                              hipStream_t stream) {
  const float* enc  = (const float*)d_in[0];
  const int*   eids = (const int*)  d_in[1];
  const float* Wih  = (const float*)d_in[2];
  const float* Whh  = (const float*)d_in[3];
  const float* bih  = (const float*)d_in[4];
  const float* bhh  = (const float*)d_in[5];
  const float* fcw  = (const float*)d_in[6];
  const float* fcb  = (const float*)d_in[7];
  float* out = (float*)d_out;

  char* p = (char*)d_ws;
  auto carve = [&](size_t bytes) -> char* {
    char* r = p; p += (bytes + 255) & ~(size_t)255; return r;
  };
  float* ts   = (float*)carve((size_t)MROWS*ENC*4);   // turn_states f32
  bf16*  tsb  = (bf16*) carve((size_t)MROWS*ENC*2);   // turn_states bf16
  bf16*  wihb = (bf16*) carve((size_t)G4*ENC*2);      // W_ih bf16
  bf16*  whhb = (bf16*) carve((size_t)G4*HID*2);      // W_hh bf16
  float* XG   = (float*)carve((size_t)MROWS*G4*4);    // precomputed x-gates
  float* ssc  = (float*)carve((size_t)NW*4);          // sscore

  pool_kernel<<<dim3(BSZ, TN), 128, 0, stream>>>(enc, eids, ts);
  f2bf_kernel<<<(MROWS*ENC/4 + 255)/256, 256, 0, stream>>>(ts, tsb, MROWS*ENC/4);
  f2bf_kernel<<<(G4*ENC/4 + 255)/256, 256, 0, stream>>>(Wih, wihb, G4*ENC/4);
  f2bf_kernel<<<(G4*HID/4 + 255)/256, 256, 0, stream>>>(Whh, whhb, G4*HID/4);
  xgemm_kernel<<<dim3(G4/16, MROWS/16), 32, 0, stream>>>(tsb, wihb, bih, bhh, XG);
  size_t lds_bytes = (size_t)(16*HID*2 + 16*G4) * sizeof(float);  // h + c + G = 192 KB
  lstm_kernel<<<NW/16, 256, lds_bytes, stream>>>(XG, whhb, fcw, ssc);
  final_kernel<<<1, 1024, 0, stream>>>(ts, fcw, fcb, ssc, out);
}
